// GlobalRNNModel_47691316855460
// MI455X (gfx1250) — matrix operand order repaired and checked by reading, not compiled, not run on hardware
//
#include <hip/hip_runtime.h>
#include <cstdint>
#include <cstddef>

typedef __attribute__((ext_vector_type(16))) __bf16 v16bf;
typedef __attribute__((ext_vector_type(8)))  __bf16 v8bf;
typedef __attribute__((ext_vector_type(8)))  float  v8f;
typedef unsigned int u32x4 __attribute__((ext_vector_type(4)));
typedef int          i32x4 __attribute__((ext_vector_type(4)));
typedef int          i32x8 __attribute__((ext_vector_type(8)));

#define BATCH    256
#define SEQLEN   48
#define HORIZON  12
#define IN_DIM   2048
#define HID      2048
#define OUT_DIM  1024
#define NGATES   (4 * HID)   /* 8192 */

// Tensor Data Mover availability (device pass only; host pass takes fallback).
#if defined(__HIP_DEVICE_COMPILE__) && __has_builtin(__builtin_amdgcn_tensor_load_to_lds) && __has_builtin(__builtin_amdgcn_s_wait_tensorcnt)
#define USE_TDM 1
#else
#define USE_TDM 0
#endif
#if __has_include(<hip/amd_detail/amd_gfx1250_TDM.h>)
#define CDNA5_TDM_6ARG 1
#endif

union BFrag { v16bf v; v8bf h[2]; };

// ---------------------------------------------------------------- conversions
__global__ void k_f32_to_bf16(const float* __restrict__ in, __bf16* __restrict__ out, long n) {
  long i = (long)blockIdx.x * blockDim.x + threadIdx.x;
  long stride = (long)gridDim.x * blockDim.x;
  for (; i < n; i += stride) out[i] = (__bf16)in[i];
}

__global__ void k_bias_sum(const float* __restrict__ a, const float* __restrict__ b,
                           float* __restrict__ out, int n) {
  int i = blockIdx.x * blockDim.x + threadIdx.x;
  if (i < n) out[i] = a[i] + b[i];
}

__global__ void k_init_state(__bf16* __restrict__ h, float* __restrict__ c, int n) {
  int i = blockIdx.x * blockDim.x + threadIdx.x;
  if (i < n) { h[i] = (__bf16)0.0f; c[i] = 0.0f; }
}

#if USE_TDM
// Issue a TDM DMA of a 64(rows) x 32(K) bf16 tile: global (row stride ldw
// elements) -> LDS row-major tile at byte offset lds_off.
// D# group0/group1 packed per CDNA5 ISA 8.3/8.4; 2-D tensor so groups 2/3 zero.
__device__ __forceinline__ void tdm_stage_tile(const __bf16* gsrc, unsigned lds_off, int ldw)
{
  unsigned long long ga = (unsigned long long)(uintptr_t)gsrc;
  u32x4 g0;
  g0[0] = 1u;                                              // count=1, user D#
  g0[1] = lds_off;                                         // lds_addr (bytes)
  g0[2] = (unsigned)ga;                                    // global_addr[31:0]
  g0[3] = (unsigned)((ga >> 32) & 0x1ffffffull)            // global_addr[56:32]
        | (2u << 30);                                      // type = 2 ("image")
  i32x8 g1;
  const unsigned td0 = (unsigned)ldw;                      // tensor_dim0 >= 32
  const unsigned td1 = 65535u;                             // tensor_dim1 large
  g1[0] = (int)(1u << 16);                                 // data_size = 2 bytes
  g1[1] = (int)((td0 & 0xffffu) << 16);                    // tensor_dim0 lo
  g1[2] = (int)((td0 >> 16) | ((td1 & 0xffffu) << 16));    // td0 hi | td1 lo
  g1[3] = (int)((td1 >> 16) | (32u << 16));                // td1 hi | tile_dim0=32
  g1[4] = (int)64;                                         // tile_dim1=64, tile_dim2=0
  g1[5] = (int)(unsigned)ldw;                              // tensor_dim0_stride lo
  g1[6] = 0;                                               // stride hi | dim1_stride lo
  g1[7] = 0;
  i32x4 z4 = {0, 0, 0, 0};
#if defined(CDNA5_TDM_6ARG)
  i32x8 z8 = {0, 0, 0, 0, 0, 0, 0, 0};
  __builtin_amdgcn_tensor_load_to_lds(g0, g1, z4, z4, z8, 0);
#else
  __builtin_amdgcn_tensor_load_to_lds(g0, g1, z4, z4, 0);
#endif
}
#endif

// ---------------------------------------------------------------- WMMA GEMM
// C[m,n] = bias[n] + sum_k A1[m,k]*W1[n,k] + sum_k A2[m,k]*W2[n,k]
// Block = 256 threads (8 wave32 waves). Block tile: 256(M) x 64(N).
// Wave w owns rows [32w, 32w+32): 2(M) x 4(N) accumulators of 16x16 f32.
// Per K-chunk(32) the 64-column weight tile (4 KB) is staged into LDS,
// double-buffered: TDM (or manual copy) of tile k+1 overlaps WMMA on tile k.
__global__ __launch_bounds__(256)
void k_gemm(const __bf16* __restrict__ A1, int lda1,
            const __bf16* __restrict__ W1, int ldw1, int K1,
            const __bf16* __restrict__ A2, int lda2,
            const __bf16* __restrict__ W2, int ldw2, int K2,
            const float* __restrict__ bias,
            float* __restrict__ C, int ldc,
            __bf16* __restrict__ Cbf, int ldcbf)
{
  __shared__ __bf16 ldsB[2][64 * 32];       // two 4 KB row-major weight tiles
  const int tid   = threadIdx.x;
  const int lane  = tid & 31;
  const int wave  = tid >> 5;
  const int nl    = lane & 15;
  const int khalf = lane >> 4;
  const int nblk  = blockIdx.x * 64;
  const int m0    = wave * 32;

  v8f acc[2][4];
#pragma unroll
  for (int mi = 0; mi < 2; ++mi)
#pragma unroll
    for (int ni = 0; ni < 4; ++ni)
#pragma unroll
      for (int r = 0; r < 8; ++r) acc[mi][ni][r] = 0.0f;

  // one K-chunk of math: A frags from global (L2), B frags from LDS tile
  auto compute = [&](const __bf16* __restrict__ A, int lda, int kk,
                     const __bf16* __restrict__ bb) {
    BFrag a[2];
#pragma unroll
    for (int mi = 0; mi < 2; ++mi) {
      const __bf16* ap = A + (size_t)(m0 + 16 * mi + nl) * lda + kk + 8 * khalf;
      a[mi].h[0] = *reinterpret_cast<const v8bf*>(ap);
      a[mi].h[1] = *reinterpret_cast<const v8bf*>(ap + 16);
    }
    BFrag b[4];
#pragma unroll
    for (int ni = 0; ni < 4; ++ni) {
      const __bf16* bp = bb + (16 * ni + nl) * 32 + 8 * khalf;
      b[ni].h[0] = *reinterpret_cast<const v8bf*>(bp);
      b[ni].h[1] = *reinterpret_cast<const v8bf*>(bp + 16);
    }
#pragma unroll
    for (int mi = 0; mi < 2; ++mi)
#pragma unroll
      for (int ni = 0; ni < 4; ++ni)
        acc[mi][ni] = __builtin_amdgcn_wmma_f32_16x16x32_bf16(
            false, a[mi].v, false, b[ni].v, (short)0, acc[mi][ni],
            false, false);
  };

#if !USE_TDM
  // manual stage: 128 threads copy 64 rows x 32 bf16 into row-major LDS tile
  auto stage_manual = [&](const __bf16* __restrict__ W, int ldw, int kk, int buf) {
    if (tid < 128) {
      const int row = tid >> 1;
      const int kh  = (tid & 1) * 16;
      const __bf16* src = W + (size_t)(nblk + row) * ldw + kk + kh;
      __bf16* dst = &ldsB[buf][row * 32 + kh];
      *reinterpret_cast<v8bf*>(dst)     = *reinterpret_cast<const v8bf*>(src);
      *reinterpret_cast<v8bf*>(dst + 8) = *reinterpret_cast<const v8bf*>(src + 8);
    }
  };
#endif

  auto mac_loop = [&](const __bf16* __restrict__ A, int lda,
                      const __bf16* __restrict__ W, int ldw, int K) {
#if USE_TDM
    const unsigned lds_off0 = (unsigned)(uintptr_t)&ldsB[0][0];
    const unsigned lds_off1 = (unsigned)(uintptr_t)&ldsB[1][0];
    __syncthreads();                           // prior tile readers done
    if (tid < 32)                              // wave 0 drives the TDM
      tdm_stage_tile(W + (size_t)nblk * ldw, lds_off0, ldw);
    int buf = 0;
    for (int kk = 0; kk < K; kk += 32) {
      if (tid < 32) {
        if (kk + 32 < K) {
          tdm_stage_tile(W + (size_t)nblk * ldw + (kk + 32),
                         buf ? lds_off0 : lds_off1, ldw);
          __builtin_amdgcn_s_wait_tensorcnt(1);   // current tile landed
        } else {
          __builtin_amdgcn_s_wait_tensorcnt(0);
        }
      }
      __syncthreads();                         // release waves on current tile
      compute(A, lda, kk, &ldsB[buf][0]);
      __syncthreads();                         // tile reads done before overwrite
      buf ^= 1;
    }
#else
    __syncthreads();                           // prior tile readers done
    stage_manual(W, ldw, 0, 0);
    int buf = 0;
    for (int kk = 0; kk < K; kk += 32) {
      __syncthreads();                         // current staged, other buf free
      if (kk + 32 < K) stage_manual(W, ldw, kk + 32, buf ^ 1);
      compute(A, lda, kk, &ldsB[buf][0]);
      buf ^= 1;
    }
#endif
  };

  if (K1 > 0) mac_loop(A1, lda1, W1, ldw1, K1);
  if (K2 > 0) mac_loop(A2, lda2, W2, ldw2, K2);

  // C/D layout: VGPR r -> row = tile_m + r + 8*(lane>>4), col = tile_n + (lane&15)
#pragma unroll
  for (int mi = 0; mi < 2; ++mi)
#pragma unroll
    for (int ni = 0; ni < 4; ++ni) {
      const int col = nblk + 16 * ni + nl;
      const float bv = bias ? bias[col] : 0.0f;
#pragma unroll
      for (int r = 0; r < 8; ++r) {
        const int row = m0 + 16 * mi + r + 8 * khalf;
        const float v = acc[mi][ni][r] + bv;
        C[(size_t)row * ldc + col] = v;
        if (Cbf) Cbf[(size_t)row * ldcbf + col] = (__bf16)v;
      }
    }
}

// ---------------------------------------------------------------- LSTM update
__global__ void k_lstm_update(const float* __restrict__ gates,
                              float* __restrict__ c, __bf16* __restrict__ h)
{
  int i = blockIdx.x * blockDim.x + threadIdx.x;
  if (i >= BATCH * HID) return;
  int b = i >> 11;                 // / HID
  int j = i & (HID - 1);
  const float* g = gates + (size_t)b * NGATES;
  float ig = g[j];
  float fg = g[HID + j];
  float gg = g[2 * HID + j];
  float og = g[3 * HID + j];
  float si = 1.0f / (1.0f + __expf(-ig));
  float sf = 1.0f / (1.0f + __expf(-fg));
  float so = 1.0f / (1.0f + __expf(-og));
  float cn = sf * c[i] + si * tanhf(gg);
  float hn = so * tanhf(cn);
  c[i] = cn;
  h[i] = (__bf16)hn;
}

// ---------------------------------------------------------------- launch
extern "C" void kernel_launch(void* const* d_in, const int* in_sizes, int n_in,
                              void* d_out, int out_size, void* d_ws, size_t ws_size,
                              hipStream_t stream)
{
  (void)in_sizes; (void)n_in; (void)out_size; (void)ws_size;

  const float* x   = (const float*)d_in[0];
  const float* Wih = (const float*)d_in[1];
  const float* Whh = (const float*)d_in[2];
  const float* bih = (const float*)d_in[3];
  const float* bhh = (const float*)d_in[4];
  const float* Wfc = (const float*)d_in[5];
  const float* bfc = (const float*)d_in[6];
  float* out = (float*)d_out;

  char* ws = (char*)d_ws;
  size_t off = 0;
  auto carve = [&](size_t bytes) -> void* {
    void* p = ws + off;
    off = (off + bytes + 255) & ~(size_t)255;
    return p;
  };
  __bf16* wih_bf = (__bf16*)carve((size_t)NGATES * IN_DIM * 2);
  __bf16* whh_bf = (__bf16*)carve((size_t)NGATES * HID * 2);
  __bf16* wfc_bf = (__bf16*)carve((size_t)OUT_DIM * HID * 2);
  __bf16* x_bf   = (__bf16*)carve((size_t)BATCH * SEQLEN * IN_DIM * 2);
  __bf16* h_bf   = (__bf16*)carve((size_t)BATCH * HID * 2);
  float*  c_st   = (float*)carve((size_t)BATCH * HID * 4);
  float*  gates  = (float*)carve((size_t)BATCH * NGATES * 4);
  float*  biasg  = (float*)carve((size_t)NGATES * 4);
  __bf16* dec_bf = (__bf16*)carve((size_t)BATCH * OUT_DIM * 2);

  // one-time (per launch) precision conversion; weights then live in L2 as bf16
  k_f32_to_bf16<<<2048, 256, 0, stream>>>(Wih, wih_bf, (long)NGATES * IN_DIM);
  k_f32_to_bf16<<<2048, 256, 0, stream>>>(Whh, whh_bf, (long)NGATES * HID);
  k_f32_to_bf16<<<2048, 256, 0, stream>>>(Wfc, wfc_bf, (long)OUT_DIM * HID);
  k_f32_to_bf16<<<2048, 256, 0, stream>>>(x,   x_bf,   (long)BATCH * SEQLEN * IN_DIM);
  k_bias_sum<<<NGATES / 256, 256, 0, stream>>>(bih, bhh, biasg, NGATES);
  k_init_state<<<(BATCH * HID) / 256, 256, 0, stream>>>(h_bf, c_st, BATCH * HID);

  // encoder: 48 steps of gates = x_t@Wih^T + h@Whh^T + (b_ih+b_hh)
  for (int t = 0; t < SEQLEN; ++t) {
    k_gemm<<<NGATES / 64, 256, 0, stream>>>(
        x_bf + (size_t)t * IN_DIM, SEQLEN * IN_DIM, wih_bf, IN_DIM, IN_DIM,
        h_bf, HID, whh_bf, HID, HID,
        biasg, gates, NGATES, (__bf16*)nullptr, 0);
    k_lstm_update<<<(BATCH * HID) / 256, 256, 0, stream>>>(gates, c_st, h_bf);
  }

  // decoder: out_t = h@Wfc^T + b_fc ; feedback input = [out_t, zeros] so only
  // the first OUT_DIM columns of W_ih contribute (K1 = 1024).
  for (int t = 0; t < HORIZON; ++t) {
    k_gemm<<<OUT_DIM / 64, 256, 0, stream>>>(
        h_bf, HID, wfc_bf, HID, HID,
        (const __bf16*)nullptr, 0, (const __bf16*)nullptr, 0, 0,
        bfc, out + (size_t)t * OUT_DIM, HORIZON * OUT_DIM,
        dec_bf, OUT_DIM);
    k_gemm<<<NGATES / 64, 256, 0, stream>>>(
        dec_bf, OUT_DIM, wih_bf, IN_DIM, OUT_DIM,
        h_bf, HID, whh_bf, HID, HID,
        biasg, gates, NGATES, (__bf16*)nullptr, 0);
    k_lstm_update<<<(BATCH * HID) / 256, 256, 0, stream>>>(gates, c_st, h_bf);
  }
}
